// TrainingModel_34505767257005
// MI455X (gfx1250) — matrix operand order repaired and checked by reading, not compiled, not run on hardware
//
#include <hip/hip_runtime.h>
#include <stdint.h>

// ---------------------------------------------------------------------------
// Hierarchical segmentation loss for MI455X (gfx1250, wave32).
//
// Structure exploited (derived from the reference's bit-packing):
//   classes 0..3  = parents, classes 4..31 = leaves, parent(l) = (l-4)/7
//   RMT[y][c] = DDESC[y][c] = (c==y) || (y>=4 && c==(y-4)/7)
//   SINK_IDS = {4,11,18,25}, sink g has parent g, siblings = other children.
//
// Pass 1 (k_main): bandwidth-bound stream over 226 MB of logits.
//   lane = voxel -> every per-channel load is fully coalesced.
//   Register accumulators: sum_p[32], ce, vw, 16 sink scalars.
//   LDS atomics (4/voxel): inter[y], inter[parent(y)], cnt[y], hist[pred,y].
// Pass 2 (k_final, 1 wave): DDESC einsums as real matmuls on v_wmma:
//   G1 = D^T H, G2 = H D via v_wmma_f32_16x16x32_f16 with an EXACT
//   hi/lo split H = 2048*Hh + Hl (both integer-exact in f16), accumulator
//   chained through the wmma C operand.
// ---------------------------------------------------------------------------

#define NUM_C 32
#define NVOX (96*96*96)
#define NB 2

// workspace accumulator layout (floats, per batch sample)
#define ACC_STRIDE 1280
#define OFF_SUMP  0
#define OFF_INTER 32
#define OFF_CNT   64
#define OFF_CE    96
#define OFF_VW    97
#define OFF_SI    98
#define OFF_SP    102
#define OFF_ST    106
#define OFF_SF    110
#define OFF_HIST  128

typedef __attribute__((ext_vector_type(16))) _Float16 v16h;
typedef __attribute__((ext_vector_type(8)))  float    v8f;

__global__ void k_zero(float* __restrict__ p, int n) {
  int i = blockIdx.x * blockDim.x + threadIdx.x;
  if (i < n) p[i] = 0.0f;
}

// ------------------------------ pass 1 -------------------------------------
__global__ __launch_bounds__(256) void k_main(
    const float* __restrict__ img, const int* __restrict__ tgt,
    const unsigned char* __restrict__ msk,
    float* __restrict__ acc, float* __restrict__ pred_out) {
  const int b = blockIdx.y;
  const float* __restrict__ im = img + (size_t)b * NUM_C * NVOX;
  const int*   __restrict__ ty = tgt + (size_t)b * NVOX;
  float* __restrict__ A  = acc + (size_t)b * ACC_STRIDE;
  float* __restrict__ po = pred_out + (size_t)b * NVOX;

  __shared__ float l_hist[NUM_C * NUM_C];
  __shared__ float l_inter[NUM_C];
  __shared__ float l_cnt[NUM_C];
  __shared__ float l_sum_p[NUM_C];
  __shared__ float l_sc[18];           // ce, vw, si[4], sp[4], st[4], sf[4]
  __shared__ float l_mf[NUM_C];
  __shared__ float l_notm[NUM_C];

  for (int i = threadIdx.x; i < NUM_C * NUM_C; i += blockDim.x) l_hist[i] = 0.f;
  if (threadIdx.x < NUM_C) {
    l_inter[threadIdx.x] = 0.f;
    l_cnt[threadIdx.x]   = 0.f;
    l_sum_p[threadIdx.x] = 0.f;
    float m = msk[b * NUM_C + threadIdx.x] ? 1.f : 0.f;
    l_mf[threadIdx.x]   = m;
    l_notm[threadIdx.x] = 1.f - m;
  }
  if (threadIdx.x < 18) l_sc[threadIdx.x] = 0.f;
  __syncthreads();

  float sum_p[NUM_C];
  #pragma unroll
  for (int c = 0; c < NUM_C; ++c) sum_p[c] = 0.f;
  float ce = 0.f, vwsum = 0.f;
  float si[4] = {0,0,0,0}, sp[4] = {0,0,0,0}, st[4] = {0,0,0,0}, sf[4] = {0,0,0,0};

  const int stride = gridDim.x * blockDim.x;
  for (int v = blockIdx.x * blockDim.x + threadIdx.x; v < NVOX; v += stride) {
    float x[NUM_C];
    #pragma unroll
    for (int c = 0; c < NUM_C; ++c) x[c] = im[(size_t)c * NVOX + v];
    if (v + stride < NVOX) __builtin_prefetch(&im[v + stride], 0, 1);

    // softmax (shared max across all channels; leaf argmax over 4..31)
    float xmax = x[0];
    #pragma unroll
    for (int c = 1; c < NUM_C; ++c) xmax = fmaxf(xmax, x[c]);
    int pr = 4; float best = x[4];
    #pragma unroll
    for (int c = 5; c < NUM_C; ++c) if (x[c] > best) { best = x[c]; pr = c; }

    float s = 0.f;
    #pragma unroll
    for (int c = 0; c < NUM_C; ++c) { float e = __expf(x[c] - xmax); x[c] = e; s += e; }
    float rs = __builtin_amdgcn_rcpf(s);
    #pragma unroll
    for (int c = 0; c < NUM_C; ++c) x[c] *= rs;

    int y = ty[v];
    float vw = (y != 255) ? 1.f : 0.f;
    int ys = (y != 255) ? y : 0;
    int yg = (ys >= 4) ? (ys - 4) / 7 : -1;   // parent class of target (also class id 0..3)

    #pragma unroll
    for (int c = 0; c < NUM_C; ++c) sum_p[c] = fmaf(x[c], vw, sum_p[c]);

    // t[c]=1 only at c=ys and c=parent(ys)
    atomicAdd(&l_inter[ys], x[ys] * vw);
    if (yg >= 0) atomicAdd(&l_inter[yg], x[yg] * vw);
    atomicAdd(&l_cnt[ys], vw);
    atomicAdd(&l_hist[pr * NUM_C + ys], vw);

    // cross-entropy: only ancestors-or-self of y contribute
    float py = fminf(fmaxf(x[ys], 1e-7f), 1.f - 1e-7f);
    float term = l_mf[ys] * __logf(py);
    if (yg >= 0) {
      float pq = fminf(fmaxf(x[yg], 1e-7f), 1.f - 1e-7f);
      term += l_mf[yg] * __logf(pq);
    }
    ce -= term * vw;
    vwsum += vw;

    // sink dice accumulators (all in registers)
    int prg = (pr - 4) / 7;
    bool pr_off = l_notm[pr] > 0.5f;
    #pragma unroll
    for (int g = 0; g < 4; ++g) {
      int sc = 4 + 7 * g;
      bool sibA = (yg == g)  && (ys != sc);
      bool sibB = (prg == g) && (pr != sc) && pr_off;
      bool sib  = sibA || sibB;
      bool pm   = (prg == g);
      float tg  = (pm && !sib) ? vw : 0.f;
      float pp  = x[sc] * vw;
      si[g] = fmaf(pp, tg, si[g]);
      sp[g] += pp;
      st[g] += tg;
      sf[g] += sib ? vw : 0.f;
    }

    po[v] = (float)pr;
  }

  // block-level reduction of register accumulators
  #pragma unroll
  for (int c = 0; c < NUM_C; ++c) atomicAdd(&l_sum_p[c], sum_p[c]);
  atomicAdd(&l_sc[0], ce);
  atomicAdd(&l_sc[1], vwsum);
  #pragma unroll
  for (int g = 0; g < 4; ++g) {
    atomicAdd(&l_sc[2  + g], si[g]);
    atomicAdd(&l_sc[6  + g], sp[g]);
    atomicAdd(&l_sc[10 + g], st[g]);
    atomicAdd(&l_sc[14 + g], sf[g]);
  }
  __syncthreads();

  for (int i = threadIdx.x; i < NUM_C * NUM_C; i += blockDim.x)
    atomicAdd(&A[OFF_HIST + i], l_hist[i]);
  if (threadIdx.x < NUM_C) {
    atomicAdd(&A[OFF_SUMP  + threadIdx.x], l_sum_p[threadIdx.x]);
    atomicAdd(&A[OFF_INTER + threadIdx.x], l_inter[threadIdx.x]);
    atomicAdd(&A[OFF_CNT   + threadIdx.x], l_cnt[threadIdx.x]);
  }
  if (threadIdx.x < 18) atomicAdd(&A[OFF_CE + threadIdx.x], l_sc[threadIdx.x]);
}

// ------------------------------ pass 2 (WMMA) ------------------------------
// 16-bit A tile 16x32 (MxK), per ISA layout: lane L holds row M=L%16,
// half h=L/16: halves i<8 -> K=8h+i, i>=8 -> K=16+8h+(i-8).
__device__ __forceinline__ v16h load_A16(const _Float16* Am, int ti, int lane, bool transA) {
  v16h a;
  int m = ti * 16 + (lane & 15);
  int h = lane >> 4;
  #pragma unroll
  for (int i = 0; i < 16; ++i) {
    int k = (i < 8) ? (8 * h + i) : (16 + 8 * h + (i - 8));
    a[i] = transA ? Am[k * 32 + m] : Am[m * 32 + k];
  }
  return a;
}
// B tile 32x16 (KxN): lane L holds column N=L%16; i<8 -> K=8h+i, i>=8 -> K=16+8h+(i-8).
__device__ __forceinline__ v16h load_B16(const _Float16* Bm, int tj, int lane) {
  v16h bv;
  int n = tj * 16 + (lane & 15);
  int h = lane >> 4;
  #pragma unroll
  for (int i = 0; i < 16; ++i) bv[i] = Bm[((i < 8) ? (8 * h + i) : (16 + 8 * h + (i - 8))) * 32 + n];
  return bv;
}
// C/D 16x16 f32: VGPR g, lanes 0-15 -> M=g, lanes 16-31 -> M=8+g, N=lane%16.
__device__ __forceinline__ void store_C32(float* Cm, v8f c, int ti, int tj, int lane) {
  int col = tj * 16 + (lane & 15);
  int h = lane >> 4;
  #pragma unroll
  for (int g = 0; g < 8; ++g) Cm[(ti * 16 + 8 * h + g) * 32 + col] = c[g];
}

// C = A * (2048*Bh + Bl)   (exact: both halves integer-exact in f16)
__device__ void mm32_splitB(const _Float16* Am, bool transA,
                            const _Float16* Bh, const _Float16* Bl,
                            float* Cm, int lane) {
  #pragma unroll
  for (int ti = 0; ti < 2; ++ti)
    #pragma unroll
    for (int tj = 0; tj < 2; ++tj) {
      v16h a  = load_A16(Am, ti, lane, transA);
      v16h bh = load_B16(Bh, tj, lane);
      v16h bl = load_B16(Bl, tj, lane);
      v8f c = {};
      c = __builtin_amdgcn_wmma_f32_16x16x32_f16(false, a, false, bh, (short)0, c, false, false);
      #pragma unroll
      for (int g = 0; g < 8; ++g) c[g] *= 2048.f;
      c = __builtin_amdgcn_wmma_f32_16x16x32_f16(false, a, false, bl, (short)0, c, false, false);
      store_C32(Cm, c, ti, tj, lane);
    }
}
// C = (2048*Ah + Al) * B
__device__ void mm32_splitA(const _Float16* Ah, const _Float16* Al,
                            const _Float16* Bm, float* Cm, int lane) {
  #pragma unroll
  for (int ti = 0; ti < 2; ++ti)
    #pragma unroll
    for (int tj = 0; tj < 2; ++tj) {
      v16h ah = load_A16(Ah, ti, lane, false);
      v16h al = load_A16(Al, ti, lane, false);
      v16h bv = load_B16(Bm, tj, lane);
      v8f c = {};
      c = __builtin_amdgcn_wmma_f32_16x16x32_f16(false, ah, false, bv, (short)0, c, false, false);
      #pragma unroll
      for (int g = 0; g < 8; ++g) c[g] *= 2048.f;
      c = __builtin_amdgcn_wmma_f32_16x16x32_f16(false, al, false, bv, (short)0, c, false, false);
      store_C32(Cm, c, ti, tj, lane);
    }
}

__global__ __launch_bounds__(32) void k_final(const float* __restrict__ acc,
                                              const unsigned char* __restrict__ msk,
                                              float* __restrict__ out) {
  __shared__ _Float16 Dm[32 * 32], Hh[32 * 32], Hl[32 * 32];
  __shared__ float G1[32 * 32], G2[32 * 32];
  const int lane = threadIdx.x;

  // DDESC row l=lane: D[l][c] = (l==c) || (l>=4 && c==(l-4)/7)
  for (int c = 0; c < 32; ++c) {
    bool d = (lane == c) || (lane >= 4 && c == (lane - 4) / 7);
    Dm[lane * 32 + c] = d ? (_Float16)1.0f : (_Float16)0.0f;
  }

  for (int b = 0; b < NB; ++b) {
    const float* A = acc + (size_t)b * ACC_STRIDE;
    for (int c = 0; c < 32; ++c) {
      float hv = A[OFF_HIST + lane * 32 + c];
      float hh = floorf(hv * (1.0f / 2048.0f));
      Hh[lane * 32 + c] = (_Float16)hh;
      Hl[lane * 32 + c] = (_Float16)(hv - hh * 2048.0f);
    }
    __syncthreads();

    mm32_splitB(Dm, /*transA=*/true, Hh, Hl, G1, lane);  // G1 = D^T * H
    mm32_splitA(Hh, Hl, Dm, G2, lane);                   // G2 = H * D
    __syncthreads();

    // confusion matrix: lane = class c
    float tp = 0.f, rsum = 0.f, csum = 0.f;
    for (int t = 0; t < 32; ++t) {
      float g1 = G1[lane * 32 + t];
      tp   += g1 * (float)Dm[t * 32 + lane];
      rsum += g1;
      csum += G2[t * 32 + lane];
    }
    float mf = msk[b * 32 + lane] ? 1.f : 0.f;
    float* cm = out + 6 + (size_t)(b * 32 + lane) * 3;
    cm[0] = tp * mf;
    cm[1] = (rsum - tp) * mf;
    cm[2] = (csum - tp) * mf;

    if (lane == 0) {
      float vw  = fmaxf(A[OFF_VW], 1.f);
      float cel = A[OFF_CE] / vw;
      float dsum = 0.f, msum = 0.f;
      for (int c = 0; c < 32; ++c) {
        float m = msk[b * 32 + c] ? 1.f : 0.f;
        float sum_t = A[OFF_CNT + c];
        if (c < 4)
          for (int i = 0; i < 7; ++i) sum_t += A[OFF_CNT + 4 + 7 * c + i];
        float dice_c = 1.f - 2.f * A[OFF_INTER + c] / (A[OFF_SUMP + c] + sum_t + 1e-5f);
        dsum += dice_c * m;
        msum += m;
      }
      float dice = dsum / fmaxf(msum, 1.f);
      float sd = 0.f, cntf = 0.f;
      for (int g = 0; g < 4; ++g) {
        float d  = 1.f - (2.f * A[OFF_SI + g] + 1e-5f) /
                         (A[OFF_SP + g] + A[OFF_ST + g] + 1e-5f);
        float fl = (A[OFF_SF + g] > 0.f) ? 1.f : 0.f;
        sd += d * fl; cntf += fl;
      }
      float sink = (cntf > 0.f) ? 0.1f * sd / fmaxf(cntf, 1.f) : 0.f;
      out[b * 3 + 0] = cel;
      out[b * 3 + 1] = dice;
      out[b * 3 + 2] = sink;
    }
    __syncthreads();
  }
}

// ------------------------------ launch -------------------------------------
extern "C" void kernel_launch(void* const* d_in, const int* in_sizes, int n_in,
                              void* d_out, int out_size, void* d_ws, size_t ws_size,
                              hipStream_t stream) {
  (void)in_sizes; (void)n_in; (void)out_size; (void)ws_size;
  const float*         images  = (const float*)d_in[0];
  const int*           targets = (const int*)d_in[1];
  const unsigned char* masks   = (const unsigned char*)d_in[2];
  float* out = (float*)d_out;
  float* acc = (float*)d_ws;

  const int accN = NB * ACC_STRIDE;
  k_zero<<<(accN + 255) / 256, 256, 0, stream>>>(acc, accN);

  // d_out layout: loss[2][3] | cm[2][32][3] | pred[2][96^3]
  float* pred_out = out + 6 + NB * NUM_C * 3;
  dim3 grid(216, NB, 1);
  k_main<<<grid, dim3(256, 1, 1), 0, stream>>>(images, targets, masks, acc, pred_out);

  k_final<<<1, 32, 0, stream>>>(acc, masks, out);
}
